// MinimalGNN_43860206026957
// MI455X (gfx1250) — compile-verified, matrix operand order checked by reading, not run on hardware
//
#include <hip/hip_runtime.h>

typedef __attribute__((ext_vector_type(16))) _Float16 v16h;
typedef __attribute__((ext_vector_type(4)))  _Float16 v4h;
typedef __attribute__((ext_vector_type(8)))  float    v8f;

#define THREADS 256

// ---------------- utility kernels ----------------

__global__ void zero_kernel(float* __restrict__ p, int count) {
  int i = blockIdx.x * THREADS + threadIdx.x;
  if (i < count) p[i] = 0.0f;
}

// Build Wcat[k][j] (f16, row-major, K x Ncol) from W_l (Ncol x halfK) and
// W_r (Ncol x halfK):  Wcat[k][j] = k < halfK ? W_l[j][k] : W_r[j][k-halfK]
__global__ void wcat_kernel(const float* __restrict__ Wl,
                            const float* __restrict__ Wr,
                            _Float16* __restrict__ out,
                            int halfK, int nShift, int total) {
  int i = blockIdx.x * THREADS + threadIdx.x;
  if (i >= total) return;
  int ncol = 1 << nShift;
  int k = i >> nShift;
  int j = i & (ncol - 1);
  float v = (k < halfK) ? Wl[j * halfK + k] : Wr[j * halfK + (k - halfK)];
  out[k * ncol + j] = (_Float16)v;
}

__global__ void degree_kernel(const long long* __restrict__ dst,
                              float* __restrict__ deg, int E) {
  int e = blockIdx.x * THREADS + threadIdx.x;
  if (e < E) {
    __hip_atomic_fetch_add(&deg[(int)dst[e]], 1.0f,
                           __ATOMIC_RELAXED, __HIP_MEMORY_SCOPE_AGENT);
  }
}

__global__ void rdeg_kernel(float* __restrict__ deg, int N) {
  int i = blockIdx.x * THREADS + threadIdx.x;
  if (i < N) deg[i] = 1.0f / fmaxf(deg[i], 1.0f);
}

// One (edge, 4-float chunk) per thread: float4 gather + 4 global f32 atomics.
__global__ void scatter_kernel(const float* __restrict__ feat,
                               float* __restrict__ sums,
                               const long long* __restrict__ src,
                               const long long* __restrict__ dst,
                               int E, int dim, int laneShift) {
  int tid = blockIdx.x * THREADS + threadIdx.x;
  int e = tid >> laneShift;
  if (e >= E) return;
  int c = (tid & ((1 << laneShift) - 1)) << 2;
  int s = (int)src[e];
  int d = (int)dst[e];
  const float4 v = *reinterpret_cast<const float4*>(feat + (size_t)s * dim + c);
  float* base = sums + (size_t)d * dim + c;
  __hip_atomic_fetch_add(base + 0, v.x, __ATOMIC_RELAXED, __HIP_MEMORY_SCOPE_AGENT);
  __hip_atomic_fetch_add(base + 1, v.y, __ATOMIC_RELAXED, __HIP_MEMORY_SCOPE_AGENT);
  __hip_atomic_fetch_add(base + 2, v.z, __ATOMIC_RELAXED, __HIP_MEMORY_SCOPE_AGENT);
  __hip_atomic_fetch_add(base + 3, v.w, __ATOMIC_RELAXED, __HIP_MEMORY_SCOPE_AGENT);
}

// ---------------- WMMA GEMM ----------------
// Load a 16-lane-wide f16 fragment per the CDNA5 16-bit A-matrix layout:
// lanes 0-15 get K pairs [off..off+7] and [off+16..off+23] (off includes the
// +8 shift for lanes 16-31). Two 16B LDS reads.
__device__ __forceinline__ v16h load_frag(const _Float16* rowBase, int off) {
  uint4 lo = *reinterpret_cast<const uint4*>(rowBase + off);
  uint4 hi = *reinterpret_cast<const uint4*>(rowBase + off + 16);
  v16h f;
  reinterpret_cast<uint4*>(&f)[0] = lo;
  reinterpret_cast<uint4*>(&f)[1] = hi;
  return f;
}

// out[N x outC] = [sum*rdeg | self] (N x KTOT) @ Wcat (KTOT x outC) + bias,
// optional ReLU.  Block: 8 waves -> 64x64 tile, wave -> 16x32 (2 WMMA accs).
// K is processed in 128-wide chunks; each chunk lies entirely in the "mean"
// half or the "self" half of K, so source selection is uniform per chunk.
template <int KTOT, bool RELU>
__global__ void __launch_bounds__(256)
sage_gemm_kernel(const float* __restrict__ sumBuf,   // N x KTOT/2
                 const float* __restrict__ selfBuf,  // N x KTOT/2
                 const float* __restrict__ rdeg,     // N (holds 1/max(deg,1))
                 const _Float16* __restrict__ Wcat,  // KTOT x outC
                 const float* __restrict__ bias,     // outC
                 float* __restrict__ out,            // N x outC
                 int N, int outC) {
  constexpr int KC   = 128;       // K chunk
  constexpr int LSTR = KC + 8;    // padded LDS row stride (f16 units)
  constexpr int KH   = KTOT / 2;
  __shared__ _Float16 As[64 * LSTR];   // A chunk, row-major [row][k]
  __shared__ _Float16 Bs[64 * LSTR];   // B chunk transposed  [n][k]

  const int tid   = threadIdx.x;
  const int lane  = tid & 31;
  const int wave  = tid >> 5;
  const int r     = wave & 3;           // row tile (0..3) within block
  const int g     = wave >> 2;          // col group (0..1) within block
  const int lanem = lane & 15;
  const int h8    = (lane >> 4) << 3;   // 0 for lanes 0-15, 8 for 16-31
  const int rowBase = blockIdx.x * 64;
  const int colBase = blockIdx.y * 64;

  v8f acc0 = {};
  v8f acc1 = {};

  for (int kc = 0; kc < KTOT; kc += KC) {
    const bool isMean = (kc < KH);                       // uniform per chunk
    const float* __restrict__ srcP = isMean ? sumBuf : selfBuf;
    const int kb = isMean ? kc : (kc - KH);

    // ---- Stage A chunk: 64 rows x 128 k, fp32 -> f16 (branchless, float4) --
    // quad index: 64 rows * 32 quads; thread handles 8 quads.
    float4 va[8];
    float  sc[8];
    int    arow[8], ak4[8];
#pragma unroll
    for (int i = 0; i < 8; ++i) {
      int idx = tid + i * 256;
      arow[i] = idx >> 5;
      ak4[i]  = (idx & 31) << 2;
      int grow = rowBase + arow[i];
      grow = (grow < N) ? grow : (N - 1);               // clamp: rows >= N discarded
      va[i] = *reinterpret_cast<const float4*>(srcP + (size_t)grow * KH + kb + ak4[i]);
      sc[i] = isMean ? rdeg[grow] : 1.0f;
    }
#pragma unroll
    for (int i = 0; i < 8; ++i) {
      v4h hv;
      hv[0] = (_Float16)(va[i].x * sc[i]);
      hv[1] = (_Float16)(va[i].y * sc[i]);
      hv[2] = (_Float16)(va[i].z * sc[i]);
      hv[3] = (_Float16)(va[i].w * sc[i]);
      *reinterpret_cast<v4h*>(&As[arow[i] * LSTR + ak4[i]]) = hv;
    }

    // ---- Stage B chunk transposed: Bs[n][k] = Wcat[kc+k][colBase+n] --------
    // quad index: 128 k * 16 quads-of-4-cols; thread handles 8 quads.
    v4h wv[8];
    int bkr[8], bj4[8];
#pragma unroll
    for (int i = 0; i < 8; ++i) {
      int q  = tid + i * 256;
      bkr[i] = q >> 4;
      bj4[i] = (q & 15) << 2;
      wv[i] = *reinterpret_cast<const v4h*>(
          Wcat + (size_t)(kc + bkr[i]) * outC + colBase + bj4[i]);
    }
#pragma unroll
    for (int i = 0; i < 8; ++i) {
#pragma unroll
      for (int t = 0; t < 4; ++t) {
        Bs[(bj4[i] + t) * LSTR + bkr[i]] = wv[i][t];
      }
    }
    __syncthreads();

    const _Float16* arowp = &As[(r * 16 + lanem) * LSTR];
    const _Float16* brow0 = &Bs[(g * 32 + lanem) * LSTR];
    const _Float16* brow1 = brow0 + 16 * LSTR;
#pragma unroll
    for (int ks = 0; ks < KC; ks += 32) {
      v16h a  = load_frag(arowp, ks + h8);
      v16h b0 = load_frag(brow0, ks + h8);
      v16h b1 = load_frag(brow1, ks + h8);
      acc0 = __builtin_amdgcn_wmma_f32_16x16x32_f16(false, a, false, b0,
                                                    (short)0, acc0, false, false);
      acc1 = __builtin_amdgcn_wmma_f32_16x16x32_f16(false, a, false, b1,
                                                    (short)0, acc1, false, false);
    }
    __syncthreads();
  }

  // Store: lane = column n, VGPR v = row (v + h8) per C/D layout.
  const int row0 = rowBase + r * 16 + h8;
#pragma unroll
  for (int t = 0; t < 2; ++t) {
    int col = colBase + g * 32 + t * 16 + lanem;
    float bv = bias[col];
    const v8f& acc = t ? acc1 : acc0;
#pragma unroll
    for (int v = 0; v < 8; ++v) {
      int row = row0 + v;
      if (row < N) {
        float val = acc[v] + bv;
        if (RELU) val = fmaxf(val, 0.0f);
        out[(size_t)row * outC + col] = val;
      }
    }
  }
}

// ---------------- launch ----------------

extern "C" void kernel_launch(void* const* d_in, const int* in_sizes, int n_in,
                              void* d_out, int out_size, void* d_ws, size_t ws_size,
                              hipStream_t stream) {
  const float*     x   = (const float*)d_in[0];
  const long long* ei  = (const long long*)d_in[1];   // int64 edge_index [2, E]
  const float*     W1l = (const float*)d_in[2];       // (256,128)
  const float*     b1  = (const float*)d_in[3];       // (256)
  const float*     W1r = (const float*)d_in[4];       // (256,128)
  const float*     W2l = (const float*)d_in[5];       // (128,256)
  const float*     b2  = (const float*)d_in[6];       // (128)
  const float*     W2r = (const float*)d_in[7];       // (128,256)
  float* out = (float*)d_out;

  const int IN = 128, HID = 256, OUT = 128;
  const int N = in_sizes[0] / IN;
  const int E = in_sizes[1] / 2;
  const long long* srcI = ei;
  const long long* dstI = ei + E;

  // Workspace layout
  float* deg  = (float*)d_ws;                      // N
  float* sum1 = deg  + N;                          // N*IN
  float* sum2 = sum1 + (size_t)N * IN;             // N*HID
  float* hbuf = sum2 + (size_t)N * HID;            // N*HID
  _Float16* W1c = (_Float16*)(hbuf + (size_t)N * HID);  // 256*256
  _Float16* W2c = W1c + 2 * IN * HID;                    // 512*128

  // 1) zero deg + sum1 + sum2 (contiguous)
  int zcount = N * (1 + IN + HID);
  zero_kernel<<<(zcount + THREADS - 1) / THREADS, THREADS, 0, stream>>>(deg, zcount);

  // 2) f16 concatenated weights
  wcat_kernel<<<(2 * IN * HID + THREADS - 1) / THREADS, THREADS, 0, stream>>>(
      W1l, W1r, W1c, IN, 8 /*256 cols*/, 2 * IN * HID);
  wcat_kernel<<<(2 * HID * OUT + THREADS - 1) / THREADS, THREADS, 0, stream>>>(
      W2l, W2r, W2c, HID, 7 /*128 cols*/, 2 * HID * OUT);

  // 3) degree -> reciprocal
  degree_kernel<<<(E + THREADS - 1) / THREADS, THREADS, 0, stream>>>(dstI, deg, E);
  rdeg_kernel<<<(N + THREADS - 1) / THREADS, THREADS, 0, stream>>>(deg, N);

  // 4) layer-1 scatter: sum1[dst] += x[src]
  {
    int total = E * (IN / 4);
    scatter_kernel<<<(total + THREADS - 1) / THREADS, THREADS, 0, stream>>>(
        x, sum1, srcI, dstI, E, IN, 5);
  }

  // 5) layer-1 GEMM (K=256): h = relu([mean|x] @ W1cat + b1)
  {
    dim3 grid((N + 63) / 64, HID / 64);
    sage_gemm_kernel<256, true><<<grid, THREADS, 0, stream>>>(
        sum1, x, deg, W1c, b1, hbuf, N, HID);
  }

  // 6) layer-2 scatter: sum2[dst] += h[src]
  {
    int total = E * (HID / 4);
    scatter_kernel<<<(total + THREADS - 1) / THREADS, THREADS, 0, stream>>>(
        hbuf, sum2, srcI, dstI, E, HID, 6);
  }

  // 7) layer-2 GEMM (K=512): out = [mean_h|h] @ W2cat + b2
  {
    dim3 grid((N + 63) / 64, OUT / 64);
    sage_gemm_kernel<512, false><<<grid, THREADS, 0, stream>>>(
        sum2, hbuf, deg, W2c, b2, out, N, OUT);
  }
}
